// SpatialConvModule_59957743452319
// MI455X (gfx1250) — compile-verified, matrix operand order checked by reading, not run on hardware
//
#include <hip/hip_runtime.h>

typedef __attribute__((ext_vector_type(16))) _Float16 v16h;
typedef __attribute__((ext_vector_type(8)))  _Float16 v8h;
typedef __attribute__((ext_vector_type(8)))  float    v8f;

#define EPS_F 1e-5f

static constexpr int    BATCH   = 8;
static constexpr int    CH      = 128;
static constexpr int    HH      = 160;
static constexpr int    WW      = 160;
static constexpr int    HP      = HH + 2;     // padded
static constexpr int    WP      = WW + 2;
static constexpr size_t X_ELEMS    = (size_t)BATCH * CH * HH * WW;   // 26,214,400
static constexpr size_t XPAD_ELEMS = (size_t)BATCH * HP * WP * CH;   // 26,873,856
static constexpr int    KDIR    = 3 * CH;     // 384
static constexpr int    KFUS    = 4 * CH;     // 512
static constexpr int    KC_DIR  = KDIR / 32;  // 12
static constexpr int    KC_FUS  = KFUS / 32;  // 16
static constexpr int    MT      = CH / 16;    // 8
static constexpr int    NWT     = WW / 16;    // 10
static constexpr int    ADIR_HALVES = MT * KC_DIR * 32 * 16;  // 49,152  (96 KB)
static constexpr int    AFUS_HALVES = MT * KC_FUS * 32 * 16;  // 65,536  (128 KB)
static constexpr int    RES_HALVES  = (WW + 2) * CH;          // 20,736  (41,472 B)

__device__ __forceinline__ v8f zero8() {
  v8f z;
  for (int i = 0; i < 8; ++i) z[i] = 0.0f;
  return z;
}

__device__ __forceinline__ v8f wmma16(v16h a, v16h b, v8f c) {
  return __builtin_amdgcn_wmma_f32_16x16x32_f16(false, a, false, b, (short)0, c, false, false);
}

__device__ __forceinline__ v16h combine(v8h lo, v8h hi) {
  v16h b;
#pragma unroll
  for (int j = 0; j < 8; ++j) { b[j] = lo[j]; b[8 + j] = hi[j]; }
  return b;
}

// A-fragment K mapping (16-bit A 16x32): lane half selects kbase 0/8;
// half j in 0..7 -> k = kbase + j ; j in 8..15 -> k = 16 + kbase + (j-8)
__device__ __forceinline__ void make_kidx(int lane, int* kidx) {
  int ko = (lane >> 4) << 3;
#pragma unroll
  for (int j = 0; j < 16; ++j) kidx[j] = (j < 8) ? (ko + j) : (16 + ko + (j - 8));
}

// copy a fragment table global -> LDS (16B granules), then barrier
__device__ __forceinline__ void stage_A(const _Float16* __restrict__ g,
                                        _Float16* __restrict__ s,
                                        int halves, int tid, int nthreads) {
  int nvec = halves >> 3;
  for (int i = tid; i < nvec; i += nthreads) ((v8h*)s)[i] = ((const v8h*)g)[i];
  __syncthreads();
}

// ------------------------------------------------ zero the padded x buffer
__global__ void k_zero(_Float16* __restrict__ p, size_t n) {
  size_t i = (size_t)blockIdx.x * blockDim.x + threadIdx.x;
  if (i < n) p[i] = (_Float16)0;
}

// ------------------------------------------------ x (NCHW f32) -> padded channel-last f16
__global__ void k_convert(const float* __restrict__ x, _Float16* __restrict__ xpad, size_t n) {
  size_t i = (size_t)blockIdx.x * blockDim.x + threadIdx.x;  // (b,h,w,c) c fastest
  if (i >= n) return;
  int c = (int)(i & (CH - 1));
  size_t pix = i >> 7;
  int w = (int)(pix % WW);
  int h = (int)((pix / WW) % HH);
  int b = (int)(pix / ((size_t)WW * HH));
  float v = x[(((size_t)b * CH + c) * HH + h) * WW + w];
  xpad[(((size_t)b * HP + (h + 1)) * WP + (w + 1)) * CH + c] = (_Float16)v;
}

// ------------------------------------------------ weight fragment prep (f16, fragment order)
__global__ void k_prep_dir(const float* __restrict__ kern, _Float16* __restrict__ wfrag) {
  int rec = blockIdx.x * blockDim.x + threadIdx.x;
  if (rec >= 4 * MT * KC_DIR * 32) return;
  int lane = rec & 31;
  int kc   = (rec >> 5) % KC_DIR;
  int mt   = (rec / (32 * KC_DIR)) % MT;
  int d    = rec / (32 * KC_DIR * MT);
  int m    = lane & 15;
  int kidx[16]; make_kidx(lane, kidx);
  int o = mt * 16 + m;
  for (int j = 0; j < 16; ++j) {
    int k = kc * 32 + kidx[j];
    int t = k >> 7, c = k & 127;
    int kh = (d < 2) ? 1 : t;
    int kw = (d < 2) ? t : 1;
    float v = kern[(((size_t)(d * CH + o) * CH + c) * 3 + kh) * 3 + kw];
    wfrag[(size_t)rec * 16 + j] = (_Float16)v;
  }
}

__global__ void k_prep_fus(const float* __restrict__ fw, _Float16* __restrict__ wfrag) {
  int rec = blockIdx.x * blockDim.x + threadIdx.x;
  if (rec >= MT * KC_FUS * 32) return;
  int lane = rec & 31;
  int kc   = (rec >> 5) % KC_FUS;
  int mt   = rec / (32 * KC_FUS);
  int m    = lane & 15;
  int kidx[16]; make_kidx(lane, kidx);
  int o = mt * 16 + m;
  for (int j = 0; j < 16; ++j) {
    int k = kc * 32 + kidx[j];
    wfrag[(size_t)rec * 16 + j] = (_Float16)fw[(size_t)o * KFUS + k];
  }
}

// ------------------------------------------------ pre = conv1d(x) + bias (channel-last out)
__global__ void __launch_bounds__(256) k_pre(
    const _Float16* __restrict__ xpad,
    const float* __restrict__ biases,
    const _Float16* __restrict__ wdir,
    _Float16* __restrict__ prefeat) {
  extern __shared__ char smem[];
  _Float16* sA = (_Float16*)smem;                         // 98,304 B
  int tid  = threadIdx.x;
  int wave = tid >> 5, lane = tid & 31;
  int d    = blockIdx.x / (BATCH * HH * NWT / 8);         // blocks-per-dir = 1600
  int sgrp = blockIdx.x % (BATCH * HH * NWT / 8);
  stage_A(wdir + (size_t)d * ADIR_HALVES, sA, ADIR_HALVES, tid, 256);
  int strip = sgrp * 8 + wave;                            // 0..12799 within direction
  int wt   = strip % NWT;
  int line = (strip / NWT) % HH;
  int b    = strip / (NWT * HH);
  bool rowdir = d < 2;
  int n     = lane & 15;
  int koff  = (lane >> 4) << 3;
  int ohalf = koff;
  int wpix  = wt * 16 + n;
  v8f acc[MT];
#pragma unroll
  for (int i = 0; i < MT; ++i) acc[i] = zero8();
  if (rowdir) {
    const _Float16* bbase =
        xpad + (((size_t)b * HP + (line + 1)) * WP + wpix) * CH + koff;
#pragma unroll
    for (int kc = 0; kc < KC_DIR; ++kc) {
      const _Float16* src = bbase + kc * 32;              // t*CH + c0 == kc*32
      v16h bf = combine(*(const v8h*)(src), *(const v8h*)(src + 16));
      v16h af[MT];
#pragma unroll
      for (int mt = 0; mt < MT; ++mt)
        af[mt] = *(const v16h*)(sA + (size_t)((mt * KC_DIR + kc) * 32 + lane) * 16);
#pragma unroll
      for (int mt = 0; mt < MT; ++mt) acc[mt] = wmma16(af[mt], bf, acc[mt]);
    }
  } else {
    const _Float16* bbase =
        xpad + (((size_t)b * HP + line) * WP + (wpix + 1)) * CH + koff;
#pragma unroll
    for (int kc = 0; kc < KC_DIR; ++kc) {
      const _Float16* src = bbase + (kc >> 2) * (WP * CH) + (kc & 3) * 32;
      v16h bf = combine(*(const v8h*)(src), *(const v8h*)(src + 16));
      v16h af[MT];
#pragma unroll
      for (int mt = 0; mt < MT; ++mt)
        af[mt] = *(const v16h*)(sA + (size_t)((mt * KC_DIR + kc) * 32 + lane) * 16);
#pragma unroll
      for (int mt = 0; mt < MT; ++mt) acc[mt] = wmma16(af[mt], bf, acc[mt]);
    }
  }
  _Float16* dst = prefeat + ((((size_t)(d * BATCH + b) * HH + line) * WW) + wpix) * CH;
#pragma unroll
  for (int mt = 0; mt < MT; ++mt) {
    int o0 = mt * 16 + ohalf;
    v8h yv;
#pragma unroll
    for (int r = 0; r < 8; ++r) yv[r] = (_Float16)(acc[mt][r] + biases[d * CH + o0 + r]);
    *(v8h*)(dst + o0) = yv;
  }
}

// ------------------------------------------------ sequential propagation (32 parallel scans)
__global__ void __launch_bounds__(320) k_scan(
    const float* __restrict__ bn_gamma, const float* __restrict__ bn_beta,
    const float* __restrict__ bn_mean,  const float* __restrict__ bn_var,
    const float* __restrict__ prelu_a,
    const _Float16* __restrict__ wdir,
    _Float16* __restrict__ prefeat) {
  extern __shared__ char smem[];
  _Float16* sA     = (_Float16*)smem;                            // 98,304 B
  _Float16* res0   = (_Float16*)(smem + ADIR_HALVES * 2);        // 2x 41,472 B [pos][c]
  _Float16* res1   = res0 + RES_HALVES;
  float*    sscale = (float*)(smem + ADIR_HALVES * 2 + 2 * RES_HALVES * 2);
  float*    sshift = sscale + CH;
  int d = blockIdx.x >> 3;
  int b = blockIdx.x & 7;
  int tid = threadIdx.x;
  for (int c = tid; c < CH; c += 320) {
    float inv = bn_gamma[d * CH + c] * rsqrtf(bn_var[d * CH + c] + EPS_F);
    sscale[c] = inv;
    sshift[c] = bn_beta[d * CH + c] - bn_mean[d * CH + c] * inv;
  }
  for (int i = tid; i < 2 * RES_HALVES; i += 320) res0[i] = (_Float16)0;
  stage_A(wdir + (size_t)d * ADIR_HALVES, sA, ADIR_HALVES, tid, 320);  // barrier inside
  float alpha = prelu_a[d];
  int wave = tid >> 5, lane = tid & 31;
  int n0 = wave * 16;
  int n  = lane & 15;
  int ng = n0 + n;
  int koff  = (lane >> 4) << 3;
  int ohalf = koff;
  bool rowdir = d < 2;
  bool rev    = (d & 1);
  _Float16* pfeat = prefeat + (size_t)(d * BATCH + b) * HH * WW * CH;
  _Float16* rcur = res0;
  _Float16* rnxt = res1;
  for (int step = 0; step < HH; ++step) {
    int pos = rev ? (HH - 1 - step) : step;
    if (step + 1 < HH) {  // pull next pre line toward caches (global_prefetch_b8)
      int npos = rev ? (pos - 1) : (pos + 1);
      int pix = n0 + lane;
      if (pix < WW) {
        size_t poff = rowdir ? ((size_t)npos * WW + pix) : ((size_t)pix * WW + npos);
        __builtin_prefetch(pfeat + poff * CH, 0, 1);
      }
    }
    v8f acc[MT];
#pragma unroll
    for (int i = 0; i < MT; ++i) acc[i] = zero8();
    const _Float16* bbase = rcur + (size_t)ng * CH + koff;  // halo: pos index ng + t
#pragma unroll
    for (int kc = 0; kc < KC_DIR; ++kc) {
      const _Float16* src = bbase + kc * 32;                // t*CH + c0 == kc*32
      v16h bf = combine(*(const v8h*)(src), *(const v8h*)(src + 16));
      v16h af[MT];
#pragma unroll
      for (int mt = 0; mt < MT; ++mt)
        af[mt] = *(const v16h*)(sA + (size_t)((mt * KC_DIR + kc) * 32 + lane) * 16);
#pragma unroll
      for (int mt = 0; mt < MT; ++mt) acc[mt] = wmma16(af[mt], bf, acc[mt]);
    }
    size_t poff = rowdir ? ((size_t)pos * WW + ng) : ((size_t)ng * WW + pos);
    _Float16* pf = pfeat + poff * CH;
#pragma unroll
    for (int mt = 0; mt < MT; ++mt) {
      int o0 = mt * 16 + ohalf;
      v8h pr = *(const v8h*)(pf + o0);
      v8f sc = *(const v8f*)(&sscale[o0]);
      v8f sh = *(const v8f*)(&sshift[o0]);
      v8h yv;
#pragma unroll
      for (int r = 0; r < 8; ++r) {
        float h = acc[mt][r] + (float)pr[r];
        float y = h * sc[r] + sh[r];
        y = (y >= 0.0f) ? y : alpha * y;
        yv[r] = (_Float16)y;
      }
      *(v8h*)(pf + o0) = yv;                                // in-place feat output
      *(v8h*)(rnxt + (size_t)(1 + ng) * CH + o0) = yv;      // next-step state
    }
    __syncthreads();  // single barrier: rnxt written, rcur reads complete
    _Float16* t2 = rcur; rcur = rnxt; rnxt = t2;
  }
}

// ------------------------------------------------ 1x1 fusion + BN + ReLU (f32 NCHW out)
__global__ void __launch_bounds__(256) k_fuse(
    const _Float16* __restrict__ prefeat,
    const _Float16* __restrict__ wfus,
    const float* __restrict__ fus_b,
    const float* __restrict__ fbn_gamma, const float* __restrict__ fbn_beta,
    const float* __restrict__ fbn_mean,  const float* __restrict__ fbn_var,
    float* __restrict__ out) {
  extern __shared__ char smem[];
  _Float16* sA     = (_Float16*)smem;                       // 131,072 B
  float*    sscale = (float*)(smem + AFUS_HALVES * 2);
  float*    sshift = sscale + CH;
  int tid  = threadIdx.x;
  int wave = tid >> 5, lane = tid & 31;
  for (int c = tid; c < CH; c += 256) {
    float inv = fbn_gamma[c] * rsqrtf(fbn_var[c] + EPS_F);
    sscale[c] = inv;
    sshift[c] = (fus_b[c] - fbn_mean[c]) * inv + fbn_beta[c];
  }
  stage_A(wfus, sA, AFUS_HALVES, tid, 256);                 // barrier inside
  int strip = blockIdx.x * 8 + wave;                        // 0..12799
  int wt = strip % NWT;
  int h  = (strip / NWT) % HH;
  int b  = strip / (NWT * HH);
  int n  = lane & 15;
  int koff  = (lane >> 4) << 3;
  int ohalf = koff;
  int wpix  = wt * 16 + n;
  v8f acc[MT];
#pragma unroll
  for (int i = 0; i < MT; ++i) acc[i] = zero8();
  const _Float16* bases[4];
#pragma unroll
  for (int df = 0; df < 4; ++df)
    bases[df] = prefeat + ((((size_t)(df * BATCH + b) * HH + h) * WW) + wpix) * CH + koff;
#pragma unroll
  for (int kc = 0; kc < KC_FUS; ++kc) {
    const _Float16* src = bases[kc >> 2] + (kc & 3) * 32;
    v16h bf = combine(*(const v8h*)(src), *(const v8h*)(src + 16));
    v16h af[MT];
#pragma unroll
    for (int mt = 0; mt < MT; ++mt)
      af[mt] = *(const v16h*)(sA + (size_t)((mt * KC_FUS + kc) * 32 + lane) * 16);
#pragma unroll
    for (int mt = 0; mt < MT; ++mt) acc[mt] = wmma16(af[mt], bf, acc[mt]);
  }
#pragma unroll
  for (int mt = 0; mt < MT; ++mt) {
    int o0 = mt * 16 + ohalf;
    v8f sc = *(const v8f*)(&sscale[o0]);
    v8f sh = *(const v8f*)(&sshift[o0]);
#pragma unroll
    for (int r = 0; r < 8; ++r) {
      float v = acc[mt][r] * sc[r] + sh[r];
      v = fmaxf(v, 0.0f);
      out[((size_t)b * CH + (o0 + r)) * HH * WW + (size_t)h * WW + wpix] = v;
    }
  }
}

// ------------------------------------------------ host launcher
extern "C" void kernel_launch(void* const* d_in, const int* in_sizes, int n_in,
                              void* d_out, int out_size, void* d_ws, size_t ws_size,
                              hipStream_t stream) {
  const float* x         = (const float*)d_in[0];
  const float* kernels   = (const float*)d_in[1];
  const float* biases    = (const float*)d_in[2];
  const float* bn_gamma  = (const float*)d_in[3];
  const float* bn_beta   = (const float*)d_in[4];
  const float* bn_mean   = (const float*)d_in[5];
  const float* bn_var    = (const float*)d_in[6];
  const float* prelu_a   = (const float*)d_in[7];
  const float* fus_w     = (const float*)d_in[8];
  const float* fus_b     = (const float*)d_in[9];
  const float* fbn_gamma = (const float*)d_in[10];
  const float* fbn_beta  = (const float*)d_in[11];
  const float* fbn_mean  = (const float*)d_in[12];
  const float* fbn_var   = (const float*)d_in[13];
  float* out = (float*)d_out;

  char* ws = (char*)d_ws;
  size_t off = 0;
  auto carve = [&](size_t bytes) -> char* {
    char* p = ws + off;
    off = (off + bytes + 255) & ~(size_t)255;
    return p;
  };
  _Float16* xpad    = (_Float16*)carve(XPAD_ELEMS * sizeof(_Float16));       // 53.7 MB
  _Float16* prefeat = (_Float16*)carve(4 * X_ELEMS * sizeof(_Float16));      // 209.7 MB
  _Float16* wdir    = (_Float16*)carve((size_t)4 * ADIR_HALVES * sizeof(_Float16));
  _Float16* wfus    = (_Float16*)carve((size_t)AFUS_HALVES * sizeof(_Float16));

  {
    int threads = 256;
    int blocks = (int)((XPAD_ELEMS + threads - 1) / threads);
    k_zero<<<blocks, threads, 0, stream>>>(xpad, XPAD_ELEMS);
  }
  {
    int threads = 256;
    int blocks = (int)((X_ELEMS + threads - 1) / threads);
    k_convert<<<blocks, threads, 0, stream>>>(x, xpad, X_ELEMS);
  }
  {
    int recs = 4 * MT * KC_DIR * 32;  // 12288
    k_prep_dir<<<(recs + 255) / 256, 256, 0, stream>>>(kernels, wdir);
  }
  {
    int recs = MT * KC_FUS * 32;      // 4096
    k_prep_fus<<<(recs + 255) / 256, 256, 0, stream>>>(fus_w, wfus);
  }
  {
    int blocks = 4 * (BATCH * HH * NWT / 8);   // 6400, each = 8 strips of one dir
    size_t shmem = (size_t)ADIR_HALVES * 2;    // 98,304 B
    k_pre<<<blocks, 256, shmem, stream>>>(xpad, biases, wdir, prefeat);
  }
  {
    size_t shmem = (size_t)ADIR_HALVES * 2 + (size_t)2 * RES_HALVES * 2 + 2 * CH * 4;  // 182,272 B
    k_scan<<<32, 320, shmem, stream>>>(bn_gamma, bn_beta, bn_mean, bn_var, prelu_a, wdir, prefeat);
  }
  {
    int blocks = BATCH * HH * NWT / 8;         // 1600
    size_t shmem = (size_t)AFUS_HALVES * 2 + 2 * CH * 4;  // 132,096 B
    k_fuse<<<blocks, 256, shmem, stream>>>(prefeat, wfus, fus_b,
                                           fbn_gamma, fbn_beta, fbn_mean, fbn_var, out);
  }
  (void)in_sizes; (void)n_in; (void)out_size; (void)ws_size;
}
